// RFP_UAGs_41420664602773
// MI455X (gfx1250) — compile-verified, matrix operand order checked by reading, not run on hardware
//
#include <hip/hip_runtime.h>

typedef __attribute__((ext_vector_type(16))) _Float16 v16h;
typedef __attribute__((ext_vector_type(8)))  float    v8f;
typedef unsigned short u16;
typedef unsigned int   u32;

namespace {

constexpr int kB = 8, kC = 64, kH = 128, kW = 128, kD = 4;
constexpr int kWavesPerBlk = 8;
constexpr int kPitchH = 68;            // LDS row pitch in halves (34 dwords)
constexpr int kPitchD = kPitchH / 2;   // 34
// x strides (elements), layout [B, C, H, W, D]
constexpr int xSW = kD, xSH = kW * kD, xSC = kH * kW * kD, xSB = kC * kH * kW * kD;

union HPack  { u32 u; _Float16 h[2]; u16 s[2]; };
union Frag16 { v16h v; u32 u[8]; };

// K striping shared by 16-bit A (16x32) and B (32x16) fragments (ISA 7.12.2):
// lanes 0-15 hold K = {0..7, 16..23}, lanes 16-31 hold K = {8..15, 24..31}.
// Per lane + k-chunk this is two contiguous 16B runs -> vectorizable loads.
__device__ __forceinline__ int kmap(int p) { return (p < 4) ? (2 * p) : (16 + 2 * (p - 4)); }

__device__ __forceinline__ v8f wmma_f16(const Frag16& a, const Frag16& b, v8f c) {
  return __builtin_amdgcn_wmma_f32_16x16x32_f16(false, a.v, false, b.v, (short)0, c,
                                                false, false);
}

// One directional scan. blockIdx.y: 0 = forward (south/east), 1 = backward (north/west).
// IN_IS_X: input is fp32 x (strided, [B,C,H,W,D] already offset by depth); scan over H,
//          sequences indexed by (b, w). Output -> hs (y=0) / hn (y=1), layout [B,H,W,C] f16.
// !IN_IS_X: input/output in-place on hs/hn ([B,H,W,C] f16); scan over W, sequences (b, h).
template <bool IN_IS_X>
__global__ __launch_bounds__(256) void uag_scan_kernel(
    const float* __restrict__ xd, u16* __restrict__ hsb, u16* __restrict__ hnb,
    const float* __restrict__ Wd, const float* __restrict__ bd,
    const float* __restrict__ gd, int sweep_base) {
  const int y = blockIdx.y;
  u16* hbuf = y ? hnb : hsb;
  const int widx = sweep_base + y * 4;                 // Wc index; Wp = widx+1
  const float g = gd[(sweep_base >> 1) + 2 * y];       // gamma: S->0 E->1 N->2 W->3
  const float* Wc = Wd + (widx + 0) * kC * kC;
  const float* Wp = Wd + (widx + 1) * kC * kC;
  const float* bc = bd + (widx + 0) * kC;
  const float* bp = bd + (widx + 1) * kC;

  const int tid = threadIdx.x;
  const int lane = tid & 31;
  const int wv = tid >> 5;
  const int waveg = blockIdx.x * kWavesPerBlk + wv;    // 0..63, each owns 16 sequences
  const int seq0 = waveg * 16;
  const int bidx = seq0 >> 7;                          // batch
  const int t0 = seq0 & 127;                           // base of inner seq index (w or h)
  const int nl = lane & 15;
  const int hi = lane >> 4;
  const int kofs = hi * 8;

  // Per-wave private LDS tile: h (16 seqs x 64 ch) f16, rows padded to 68 halves so
  // fragment reads are 8B-aligned ds_load_b64 with distinct banks per row.
  __shared__ u16 HB[kWavesPerBlk * 16 * kPitchH];
  u16* hb = HB + wv * (16 * kPitchH);
  const u32* hbu = (const u32*)hb;

  // ---- resident weight B-fragments (B[k][n] = W[n][k]) + fused per-lane biases ----
  Frag16 Bc4[4][2], Bp4[4][2];
  float bfold[4];
#pragma unroll
  for (int t2 = 0; t2 < 4; ++t2) {
    const int n = t2 * 16 + nl;
    bfold[t2] = bc[n] + g * bp[n];     // relu(cur + bc + g*(rec + bp)) = relu(cur + g*rec + bfold)
#pragma unroll
    for (int kc = 0; kc < 2; ++kc) {
#pragma unroll
      for (int p = 0; p < 8; ++p) {
        const int k = kc * 32 + kmap(p) + kofs;
        HPack pk;
        pk.h[0] = (_Float16)Wc[n * kC + k];
        pk.h[1] = (_Float16)Wc[n * kC + k + 1];
        Bc4[t2][kc].u[p] = pk.u;
        pk.h[0] = (_Float16)Wp[n * kC + k];
        pk.h[1] = (_Float16)Wp[n * kC + k + 1];
        Bp4[t2][kc].u[p] = pk.u;
      }
    }
  }

  // Coalesced f16 row store: lane writes row mr=nl, halves [hi*32, hi*32+32).
  auto store_row = [&](int pos) {
    const int mr = nl;
    int obase;
    if (IN_IS_X) obase = ((bidx * kH + pos) * kW + (t0 + mr)) * kC;   // [b][h=pos][w][c]
    else         obase = ((bidx * kH + (t0 + mr)) * kW + pos) * kC;   // [b][h][w=pos][c]
    uint4* outv = (uint4*)hbuf;
    const int vbase = (obase + hi * 32) >> 3;  // uint4 = 8 halves
#pragma unroll
    for (int j = 0; j < 4; ++j) {
      const u32* s = hbu + mr * kPitchD + hi * 16 + j * 4;  // 8B aligned
      uint2 a = *(const uint2*)(s);
      uint2 b2 = *(const uint2*)(s + 2);
      uint4 q{a.x, a.y, b2.x, b2.y};
      outv[vbase + j] = q;
    }
  };

  // ---- step 0: h0 = relu(In_0), written in D-layout coordinates ----
  const int pos0 = y ? 127 : 0;
#pragma unroll
  for (int t2 = 0; t2 < 4; ++t2) {
#pragma unroll
    for (int r = 0; r < 8; ++r) {
      const int mrow = r + hi * 8;
      const int n = t2 * 16 + nl;
      float v;
      if (IN_IS_X) {
        v = xd[bidx * xSB + n * xSC + pos0 * xSH + (t0 + mrow) * xSW];
      } else {
        const int hidx = ((bidx * kH + (t0 + mrow)) * kW + pos0) * kC + n;
        v = (float)((const _Float16*)hbuf)[hidx];
      }
      v = fmaxf(v, 0.0f);
      HPack pk;
      pk.h[0] = (_Float16)v;
      hb[mrow * kPitchH + n] = pk.s[0];
    }
  }
  asm volatile("s_wait_dscnt 0x0" ::: "memory");  // per-wave LDS RAW fence
  store_row(pos0);

  const int xrow = IN_IS_X ? (bidx * xSB + (t0 + nl) * xSW) : 0;
  const int inb  = IN_IS_X ? 0 : ((bidx * kH + (t0 + nl)) * kW) * kC;  // half units

  // ---- 127 recurrent steps: h = relu(Wc*in + g*Wp*h_prev + bfold) ----
  for (int i = 1; i < 128; ++i) {
    const int pos = y ? (127 - i) : i;

    // A-fragments of h_prev from LDS: 4x ds_load_b64 per k-chunk
    Frag16 Ah[2];
#pragma unroll
    for (int kc = 0; kc < 2; ++kc) {
      const u32* s = hbu + nl * kPitchD + kc * 16 + hi * 4;
      uint2 q;
      q = *(const uint2*)(s + 0);  Ah[kc].u[0] = q.x; Ah[kc].u[1] = q.y;
      q = *(const uint2*)(s + 2);  Ah[kc].u[2] = q.x; Ah[kc].u[3] = q.y;
      q = *(const uint2*)(s + 8);  Ah[kc].u[4] = q.x; Ah[kc].u[5] = q.y;
      q = *(const uint2*)(s + 10); Ah[kc].u[6] = q.x; Ah[kc].u[7] = q.y;
    }

    // A-fragments of the current input
    Frag16 Ain[2];
    if (IN_IS_X) {
      const int base_cm = xrow + pos * xSH;   // fp32, channel stride 65536 -> scalar gathers
#pragma unroll
      for (int kc = 0; kc < 2; ++kc)
#pragma unroll
        for (int p = 0; p < 8; ++p) {
          const int k = kc * 32 + kmap(p) + kofs;
          HPack pk;
          pk.h[0] = (_Float16)xd[base_cm + k * xSC];
          pk.h[1] = (_Float16)xd[base_cm + (k + 1) * xSC];
          Ain[kc].u[p] = pk.u;
        }
      if (i < 127)  // global_prefetch_b8 of next scan row of x
        __builtin_prefetch(xd + xrow + (y ? (pos - 1) : (pos + 1)) * xSH, 0, 1);
    } else {
      const u32* inu = (const u32*)hbuf;
      const int rb = (inb + pos * kC) >> 1;   // dword base of this lane's row, 128B aligned
#pragma unroll
      for (int kc = 0; kc < 2; ++kc) {
        const uint4* inv = (const uint4*)(inu + rb + kc * 16 + hi * 4);  // 16B aligned
        uint4 q0 = inv[0];            // halves kc*32+kofs .. +7
        uint4 q1 = inv[2];            // halves kc*32+16+kofs .. +7
        Ain[kc].u[0] = q0.x; Ain[kc].u[1] = q0.y; Ain[kc].u[2] = q0.z; Ain[kc].u[3] = q0.w;
        Ain[kc].u[4] = q1.x; Ain[kc].u[5] = q1.y; Ain[kc].u[6] = q1.z; Ain[kc].u[7] = q1.w;
      }
    }

    // 4 N-tiles x (2 K-chunks x 2 GEMMs) = 16 v_wmma per step, fp32 accumulate
#pragma unroll
    for (int t2 = 0; t2 < 4; ++t2) {
      v8f accc = {0.f, 0.f, 0.f, 0.f, 0.f, 0.f, 0.f, 0.f};
      accc = wmma_f16(Ain[0], Bc4[t2][0], accc);
      accc = wmma_f16(Ain[1], Bc4[t2][1], accc);
      v8f accr = {0.f, 0.f, 0.f, 0.f, 0.f, 0.f, 0.f, 0.f};
      accr = wmma_f16(Ah[0], Bp4[t2][0], accr);
      accr = wmma_f16(Ah[1], Bp4[t2][1], accr);
#pragma unroll
      for (int r = 0; r < 8; ++r) {
        float hv = fmaf(g, accr[r], accc[r]) + bfold[t2];
        hv = fmaxf(hv, 0.0f);
        HPack pk;
        pk.h[0] = (_Float16)hv;
        hb[(r + hi * 8) * kPitchH + t2 * 16 + nl] = pk.s[0];
      }
    }
    asm volatile("s_wait_dscnt 0x0" ::: "memory");
    store_row(pos);
  }
}

// out[b,c,h,w,d] = x + hse + hnw for one depth slice
__global__ __launch_bounds__(256) void uag_final_kernel(
    const float* __restrict__ x, const u16* __restrict__ hs,
    const u16* __restrict__ hn, float* __restrict__ out, int d) {
  const int id = blockIdx.x * 256 + threadIdx.x;        // (b,c,h,w) flat
  const int w = id & 127, h = (id >> 7) & 127, c = (id >> 14) & 63, b = id >> 20;
  const int xi = id * kD + d;
  const int hidx = ((b * kH + h) * kW + w) * kC + c;
  out[xi] = x[xi] + (float)((const _Float16*)hs)[hidx] + (float)((const _Float16*)hn)[hidx];
}

}  // namespace

extern "C" void kernel_launch(void* const* d_in, const int* in_sizes, int n_in,
                              void* d_out, int out_size, void* d_ws, size_t ws_size,
                              hipStream_t stream) {
  (void)in_sizes; (void)n_in; (void)out_size; (void)ws_size;
  const float* x  = (const float*)d_in[0];
  const float* Wt = (const float*)d_in[1];
  const float* bt = (const float*)d_in[2];
  const float* gt = (const float*)d_in[3];
  float* out = (float*)d_out;

  // workspace: two f16 [B,H,W,C] buffers (16.78 MB each) reused across depths
  u16* hs = (u16*)d_ws;
  u16* hn = hs + (size_t)kB * kH * kW * kC;

  const dim3 blk(256, 1, 1);
  const dim3 grdS(8, 2, 1);   // 8 blocks x 8 waves x 16 seqs = 1024 sequences; y = fwd/bwd
  const dim3 grdF((kB * kC * kH * kW) / 256, 1, 1);

  for (int d = 0; d < kD; ++d) {
    const float* xd = x + d;
    const float* Wd = Wt + (size_t)d * 8 * kC * kC;
    const float* bd = bt + (size_t)d * 8 * kC;
    const float* gd = gt + (size_t)d * 4;
    // south (y=0) + north (y=1): read x, write hs / hn
    uag_scan_kernel<true><<<grdS, blk, 0, stream>>>(xd, hs, hn, Wd, bd, gd, 0);
    // east (y=0, in-place on hs) + west (y=1, in-place on hn)
    uag_scan_kernel<false><<<grdS, blk, 0, stream>>>(xd, hs, hn, Wd, bd, gd, 2);
    // residual combine for this depth
    uag_final_kernel<<<grdF, blk, 0, stream>>>(x, hs, hn, out, d);
  }
}
